// PointMLPSemSegModel_57853209477794
// MI455X (gfx1250) — matrix operand order repaired and checked by reading, not compiled, not run on hardware
//
#include <hip/hip_runtime.h>
#include <math.h>

typedef __attribute__((ext_vector_type(16))) _Float16 v16h;
typedef __attribute__((ext_vector_type(8)))  _Float16 h8;
typedef __attribute__((ext_vector_type(4)))  _Float16 h4;
typedef __attribute__((ext_vector_type(2)))  _Float16 h2;
typedef __attribute__((ext_vector_type(8)))  float    v8f;
typedef __attribute__((ext_vector_type(4)))  float    f4;

#define KNN 24

// ------------------------------------------------------------------
// WMMA GEMM:  Y[b] = W (M x Kd)  *  X[b] (Kd x Ncols)  (+ bias, EDGE only)
// f32 in/out, f16 WMMA compute, f32 accumulate.
// Block = 64 threads = 2 waves; each wave owns one 16x16 C tile.
// FAST (EDGE=false): requires Kd%32==0, M%16==0, ld%32==0 — no guards,
//   vectorized b128 global loads, packed LDS stores, b128 fragment reads.
// EDGE (EDGE=true):  fully guarded scalar staging (embed K=9, cls M=13).
// ------------------------------------------------------------------
template <bool EDGE>
__global__ __launch_bounds__(64)
void k_gemm(const float* __restrict__ W, const float* __restrict__ X,
            float* __restrict__ Y, const float* __restrict__ bias,
            int M, int Kd, int Ncols, int ldx, int ldy)
{
    __shared__ _Float16 Wl[16][40];   // [row m][k]   16x32 A tile (b128-aligned rows)
    __shared__ _Float16 Xl[32][40];   // [col n][k]   32x32 B tile, n-major

    const int b       = blockIdx.z;
    const int mBase   = blockIdx.y * 16;
    const int colBase = blockIdx.x * 32;
    const float* Xb = X + (size_t)b * Kd * ldx;
    float*       Yb = Y + (size_t)b * M  * ldy;

    const int tid  = threadIdx.x;
    const int wv   = tid >> 5;
    const int lane = tid & 31;
    const int mrow = lane & 15;
    const int hi   = lane >> 4;
    const int nloc = (lane & 15) + wv * 16;

    v8f acc = {};

    for (int k0 = 0; k0 < Kd; k0 += 32) {
        if (EDGE) {
            // guarded scalar staging (rare shapes only)
            for (int e = tid; e < 512; e += 64) {
                int r = e >> 5, kk = e & 31;
                float v = 0.f;
                if ((mBase + r) < M && (k0 + kk) < Kd)
                    v = W[(size_t)(mBase + r) * Kd + k0 + kk];
                Wl[r][kk] = (_Float16)v;
            }
            for (int e = tid; e < 1024; e += 64) {
                int kk = e >> 5, cc = e & 31;
                float v = 0.f;
                if ((k0 + kk) < Kd && (colBase + cc) < Ncols)
                    v = Xb[(size_t)(k0 + kk) * ldx + colBase + cc];
                Xl[cc][kk] = (_Float16)v;
            }
        } else {
            // W tile: thread t -> row r = t/4, 8 consecutive k
            {
                int r  = tid >> 2;
                int kk = (tid & 3) * 8;
                const float* src = W + (size_t)(mBase + r) * Kd + k0 + kk;
                f4 v0 = *(const f4*)(src);
                f4 v1 = *(const f4*)(src + 4);
                h4 p0, p1;
#pragma unroll
                for (int j = 0; j < 4; ++j) { p0[j] = (_Float16)v0[j]; p1[j] = (_Float16)v1[j]; }
                *(h4*)&Wl[r][kk]     = p0;
                *(h4*)&Wl[r][kk + 4] = p1;
            }
            // X tile: 2 patches of (2 k-rows x 4 cols), transpose in registers
#pragma unroll
            for (int pq = 0; pq < 2; ++pq) {
                int id  = tid * 2 + pq;
                int kp  = id >> 3;           // k pair index 0..15
                int ccq = (id & 7) * 4;      // col quad 0..28
                const float* src = Xb + (size_t)(k0 + 2 * kp) * ldx + colBase + ccq;
                f4 r0 = *(const f4*)(src);
                f4 r1 = *(const f4*)(src + ldx);
#pragma unroll
                for (int j = 0; j < 4; ++j) {
                    h2 pr; pr[0] = (_Float16)r0[j]; pr[1] = (_Float16)r1[j];
                    *(h2*)&Xl[ccq + j][2 * kp] = pr;
                }
            }
            if (k0 + 32 < Kd)   // hint next X k-tile into cache
                __builtin_prefetch(Xb + (size_t)(k0 + 32) * ldx + colBase + lane, 0, 1);
        }
        __syncthreads();

        // fragments per CDNA5 16-bit A/B VGPR layouts (b128 LDS reads)
        h8 a0 = *(const h8*)&Wl[mrow][hi * 8];
        h8 a1 = *(const h8*)&Wl[mrow][16 + hi * 8];
        h8 b0 = *(const h8*)&Xl[nloc][hi * 8];
        h8 b1 = *(const h8*)&Xl[nloc][16 + hi * 8];
        v16h a  = __builtin_shufflevector(a0, a1, 0,1,2,3,4,5,6,7,8,9,10,11,12,13,14,15);
        v16h bf = __builtin_shufflevector(b0, b1, 0,1,2,3,4,5,6,7,8,9,10,11,12,13,14,15);

        acc = __builtin_amdgcn_wmma_f32_16x16x32_f16(
                  false, a, false, bf, (short)0, acc, false, false);
        __syncthreads();
    }

    const int col = colBase + nloc;
    if (EDGE) {
        if (col < Ncols) {
#pragma unroll
            for (int v = 0; v < 8; ++v) {
                int mo = mBase + hi * 8 + v;
                if (mo < M)
                    Yb[(size_t)mo * ldy + col] = acc[v] + (bias ? bias[mo] : 0.f);
            }
        }
    } else {
        float* dst = Yb + (size_t)(mBase + hi * 8) * ldy + col;
#pragma unroll
        for (int v = 0; v < 8; ++v) dst[(size_t)v * ldy] = acc[v];
    }
}

// ------------------------------------------------------------------
// BatchNorm pass 1: per-channel mean/var over (batch, logical cols)
// ------------------------------------------------------------------
__global__ __launch_bounds__(256)
void k_bn_stats(const float* __restrict__ Y, float* __restrict__ mean,
                float* __restrict__ var, int M, int Ncols, int ld, int nb)
{
    const int m = blockIdx.x, tid = threadIdx.x;
    const int total = nb * Ncols;
    float s1 = 0.f, s2 = 0.f;
    for (int i = tid; i < total; i += 256) {
        int bi = i / Ncols, c = i - bi * Ncols;
        float v = Y[((size_t)bi * M + m) * ld + c];
        s1 += v; s2 += v * v;
    }
    __shared__ float r1[256], r2[256];
    r1[tid] = s1; r2[tid] = s2;
    __syncthreads();
    for (int s = 128; s > 0; s >>= 1) {
        if (tid < s) { r1[tid] += r1[tid + s]; r2[tid] += r2[tid + s]; }
        __syncthreads();
    }
    if (tid == 0) {
        float T = (float)total, mu = r1[0] / T;
        mean[m] = mu;
        var[m]  = fmaxf(r2[0] / T - mu * mu, 0.f);
    }
}

// BN pass 2: normalize * gamma + beta (+ residual) (+ ReLU)
__global__ void k_bn_apply(float* __restrict__ out, const float* __restrict__ Yr,
                           const float* __restrict__ res,
                           const float* __restrict__ mean, const float* __restrict__ var,
                           const float* __restrict__ gam, const float* __restrict__ bet,
                           int M, int Ncols, int ld, int total, int relu)
{
    int i = blockIdx.x * 256 + threadIdx.x;
    if (i >= total) return;
    int c  = i % Ncols;
    int m  = (i / Ncols) % M;
    int bi = i / (Ncols * M);
    size_t a = ((size_t)bi * M + m) * ld + c;
    float v = (Yr[a] - mean[m]) * rsqrtf(var[m] + 1e-5f) * gam[m] + bet[m];
    if (res)  v += res[a];
    if (relu) v = fmaxf(v, 0.f);
    out[a] = v;
}

// ------------------------------------------------------------------
// Farthest point sampling: one workgroup per batch, distances in LDS
// ------------------------------------------------------------------
__global__ __launch_bounds__(256)
void k_fps(const float* __restrict__ xyz, int* __restrict__ idxOut,
           int N, int npoint)
{
    __shared__ float dist[4096];
    __shared__ float rv[256];
    __shared__ int   ri[256];
    __shared__ int   sfar;
    const int b = blockIdx.x, tid = threadIdx.x;
    const float* P = xyz + (size_t)b * N * 3;
    for (int n = tid; n < N; n += 256) dist[n] = 1e10f;
    if (tid == 0) sfar = 0;
    __syncthreads();
    for (int i = 0; i < npoint; ++i) {
        int far = sfar;
        if (tid == 0) idxOut[b * npoint + i] = far;
        float cx = P[far * 3], cy = P[far * 3 + 1], cz = P[far * 3 + 2];
        float bm = -1.f; int ba = 0;
        for (int n = tid; n < N; n += 256) {
            float dx = P[n * 3] - cx, dy = P[n * 3 + 1] - cy, dz = P[n * 3 + 2] - cz;
            float d = dx * dx + dy * dy + dz * dz;
            float nd = fminf(dist[n], d);
            dist[n] = nd;
            if (nd > bm) { bm = nd; ba = n; }
        }
        rv[tid] = bm; ri[tid] = ba;
        __syncthreads();
        for (int s = 128; s > 0; s >>= 1) {
            if (tid < s && rv[tid + s] > rv[tid]) { rv[tid] = rv[tid + s]; ri[tid] = ri[tid + s]; }
            __syncthreads();
        }
        if (tid == 0) sfar = ri[0];
        __syncthreads();
    }
}

// ------------------------------------------------------------------
// misc elementwise / gather kernels
// ------------------------------------------------------------------
__global__ void k_zero(float* p, int n)
{
    int i = blockIdx.x * blockDim.x + threadIdx.x;
    if (i < n) p[i] = 0.f;
}

__global__ void k_xyz0(const float* __restrict__ x, float* __restrict__ out, int total)
{   // x (B,9,4096) -> xyz (B,4096,3)
    int i = blockIdx.x * 256 + threadIdx.x;
    if (i >= total) return;
    int c = i % 3, n = (i / 3) % 4096, b = i / (3 * 4096);
    out[i] = x[((size_t)b * 9 + c) * 4096 + n];
}

__global__ void k_gather_xyz(const float* __restrict__ xyz, const int* __restrict__ fidx,
                             float* __restrict__ out, int N, int S, int total)
{
    int i = blockIdx.x * 256 + threadIdx.x;
    if (i >= total) return;
    int c = i % 3, s = (i / 3) % S, b = i / (3 * S);
    out[i] = xyz[((size_t)b * N + fidx[b * S + s]) * 3 + c];
}

__global__ void k_gather_pts(const float* __restrict__ feats, const int* __restrict__ fidx,
                             float* __restrict__ out, int N, int S, int D, int total)
{   // feats (B,D,N) -> out (B,S,D)
    int i = blockIdx.x * 256 + threadIdx.x;
    if (i >= total) return;
    int d = i % D, s = (i / D) % S, b = i / (D * S);
    out[i] = feats[((size_t)b * D + d) * N + fidx[b * S + s]];
}

__global__ void k_knn(const float* __restrict__ q, const float* __restrict__ xyz,
                      int* __restrict__ out, int N, int S, int total)
{
    int i = blockIdx.x * 64 + threadIdx.x;   // i = b*S + s
    if (i >= total) return;
    int b = i / S;
    float qx = q[i * 3], qy = q[i * 3 + 1], qz = q[i * 3 + 2];
    const float* P = xyz + (size_t)b * N * 3;
    float bd[KNN]; int bi[KNN];
    for (int k = 0; k < KNN; ++k) { bd[k] = 3e38f; bi[k] = 0; }
    for (int n = 0; n < N; ++n) {
        float dx = P[n * 3] - qx, dy = P[n * 3 + 1] - qy, dz = P[n * 3 + 2] - qz;
        float d = dx * dx + dy * dy + dz * dz;
        if (d < bd[KNN - 1]) {
            int j = KNN - 1;
            while (j > 0 && bd[j - 1] > d) { bd[j] = bd[j - 1]; bi[j] = bi[j - 1]; --j; }
            bd[j] = d; bi[j] = n;
        }
    }
    for (int k = 0; k < KNN; ++k) out[(size_t)i * KNN + k] = bi[k];
}

__global__ __launch_bounds__(256)
void k_group_stats(const float* __restrict__ feats, const float* __restrict__ mp,
                   const int* __restrict__ knn, float* __restrict__ stats,
                   int N, int S, int D)
{
    const int b = blockIdx.y, tid = threadIdx.x;
    const int per = S * KNN * D;
    float s1 = 0.f, s2 = 0.f;
    for (int i = blockIdx.x * 256 + tid; i < per; i += gridDim.x * 256) {
        int d = i % D, k = (i / D) % KNN, s = i / (D * KNN);
        float diff = feats[((size_t)b * D + d) * N + knn[((size_t)b * S + s) * KNN + k]]
                   - mp[((size_t)b * S + s) * D + d];
        s1 += diff; s2 += diff * diff;
    }
    __shared__ float r1[256], r2[256];
    r1[tid] = s1; r2[tid] = s2;
    __syncthreads();
    for (int s = 128; s > 0; s >>= 1) {
        if (tid < s) { r1[tid] += r1[tid + s]; r2[tid] += r2[tid + s]; }
        __syncthreads();
    }
    if (tid == 0) {
        atomicAdd(&stats[2 * b],     r1[0]);
        atomicAdd(&stats[2 * b + 1], r2[0]);
    }
}

__global__ void k_group_write(const float* __restrict__ feats, const float* __restrict__ mp,
                              const int* __restrict__ knn, const float* __restrict__ stats,
                              const float* __restrict__ alpha, const float* __restrict__ beta,
                              float* __restrict__ out, int N, int S, int D, int total)
{   // out (B*S, 2D, 32-padded): [0,D) = alpha*norm+beta, [D,2D) = mean
    int i = blockIdx.x * 256 + threadIdx.x;
    if (i >= total) return;
    int d = i % D, k = (i / D) % KNN, s = (i / (D * KNN)) % S, b = i / (D * KNN * S);
    float s1 = stats[2 * b], s2 = stats[2 * b + 1];
    float T  = (float)(S * KNN * D);
    float vr = (s2 - s1 * s1 / T) / (T - 1.f);
    float sd = sqrtf(fmaxf(vr, 0.f));
    float mn = mp[((size_t)b * S + s) * D + d];
    float df = feats[((size_t)b * D + d) * N + knn[((size_t)b * S + s) * KNN + k]] - mn;
    float g  = alpha[d] * (df / (sd + 1e-5f)) + beta[d];
    size_t base = (size_t)(b * S + s) * (2 * D);
    out[(base + d)     * 32 + k] = g;
    out[(base + D + d) * 32 + k] = mn;
}

__global__ void k_maxpool(const float* __restrict__ in, float* __restrict__ out,
                          int S, int C, int total)
{   // in (B*S, C, 32-padded; 24 valid) -> out (B, C, S)
    int i = blockIdx.x * 256 + threadIdx.x;
    if (i >= total) return;
    int c = i % C, s = (i / C) % S, b = i / (C * S);
    const float* p = in + ((size_t)(b * S + s) * C + c) * 32;
    float m = -3e38f;
    for (int k = 0; k < KNN; ++k) m = fmaxf(m, p[k]);
    out[((size_t)b * C + c) * S + s] = m;
}

__global__ void k_nn3(const float* __restrict__ xyz1, const float* __restrict__ xyz2,
                      int* __restrict__ idx3, float* __restrict__ d3,
                      int N1, int N2, int total)
{
    int i = blockIdx.x * 64 + threadIdx.x;   // i = b*N1 + n
    if (i >= total) return;
    int b = i / N1;
    float qx = xyz1[i * 3], qy = xyz1[i * 3 + 1], qz = xyz1[i * 3 + 2];
    const float* P = xyz2 + (size_t)b * N2 * 3;
    float d0 = 3e38f, d1 = 3e38f, d2 = 3e38f;
    int   i0 = 0, i1 = 0, i2 = 0;
    for (int n = 0; n < N2; ++n) {
        float dx = P[n * 3] - qx, dy = P[n * 3 + 1] - qy, dz = P[n * 3 + 2] - qz;
        float d = dx * dx + dy * dy + dz * dz;
        if (d < d0)      { d2 = d1; i2 = i1; d1 = d0; i1 = i0; d0 = d; i0 = n; }
        else if (d < d1) { d2 = d1; i2 = i1; d1 = d; i1 = n; }
        else if (d < d2) { d2 = d; i2 = n; }
    }
    idx3[i * 3] = i0; idx3[i * 3 + 1] = i1; idx3[i * 3 + 2] = i2;
    d3[i * 3] = d0;   d3[i * 3 + 1] = d1;   d3[i * 3 + 2] = d2;
}

__global__ void k_copychan(const float* __restrict__ pts1, float* __restrict__ cat,
                           int C1, int Ct, int N1, int total)
{
    int i = blockIdx.x * 256 + threadIdx.x;
    if (i >= total) return;
    int n = i % N1, c = (i / N1) % C1, b = i / (N1 * C1);
    cat[((size_t)b * Ct + c) * N1 + n] = pts1[((size_t)b * C1 + c) * N1 + n];
}

__global__ void k_interp(const float* __restrict__ pts2, const int* __restrict__ idx3,
                         const float* __restrict__ d3, float* __restrict__ cat,
                         int C1, int C2, int Ct, int N1, int N2, int total)
{
    int i = blockIdx.x * 256 + threadIdx.x;
    if (i >= total) return;
    int n = i % N1, c = (i / N1) % C2, b = i / (N1 * C2);
    int base = (b * N1 + n) * 3;
    float w0 = 1.f / (d3[base] + 1e-8f);
    float w1 = 1.f / (d3[base + 1] + 1e-8f);
    float w2 = 1.f / (d3[base + 2] + 1e-8f);
    float ws = w0 + w1 + w2;
    const float* P2 = pts2 + ((size_t)b * C2 + c) * N2;
    float v = (w0 * P2[idx3[base]] + w1 * P2[idx3[base + 1]] + w2 * P2[idx3[base + 2]]) / ws;
    cat[((size_t)b * Ct + C1 + c) * N1 + n] = v;
}

// ------------------------------------------------------------------
// host orchestration
// ------------------------------------------------------------------
struct ConvP { const float* w; const float* g; const float* b; };
struct ResP  { ConvP n1, n2; };
struct StageP { const float* alpha; const float* beta; ConvP transfer; ResP pre[2]; ResP pos[2]; };
struct FpP   { ConvP fuse; ResP ext; };

extern "C" void kernel_launch(void* const* d_in, const int* in_sizes, int n_in,
                              void* d_out, int out_size, void* d_ws, size_t ws_size,
                              hipStream_t stream)
{
    (void)in_sizes; (void)n_in; (void)out_size; (void)ws_size;

    const int B = 4;
    const int Ns[5] = {4096, 2048, 1024, 512, 256};
    const int Cs[5] = {64, 128, 256, 512, 1024};

    // ---- parse params (insertion-order flatten of setup_inputs dict) ----
    int p = 0;
    auto nxt = [&]() { return (const float*)d_in[p++]; };
    auto getConv = [&]() { ConvP c; c.w = nxt(); c.g = nxt(); c.b = nxt(); return c; };
    auto getRes  = [&]() { ResP r; r.n1 = getConv(); r.n2 = getConv(); return r; };

    const float* x = nxt();
    ConvP embed = getConv();
    StageP st[4];
    for (int i = 0; i < 4; ++i) {
        st[i].alpha = nxt(); st[i].beta = nxt();
        st[i].transfer = getConv();
        st[i].pre[0] = getRes(); st[i].pre[1] = getRes();
        st[i].pos[0] = getRes(); st[i].pos[1] = getRes();
    }
    FpP fp[4];
    for (int j = 0; j < 4; ++j) { fp[j].fuse = getConv(); fp[j].ext = getRes(); }
    ConvP cls1 = getConv();
    const float* clsW = nxt();
    const float* clsB = nxt();

    // ---- workspace bump allocator ----
    size_t off = 0;
    auto allocf = [&](size_t n) {
        float* r = (float*)((char*)d_ws + off);
        off += ((n * sizeof(float)) + 255) / 256 * 256;
        return r;
    };
    auto alloci = [&](size_t n) {
        int* r = (int*)((char*)d_ws + off);
        off += ((n * sizeof(int)) + 255) / 256 * 256;
        return r;
    };

    float* xyzb[5];  for (int l = 0; l < 5; ++l) xyzb[l] = allocf((size_t)B * Ns[l] * 3);
    float* featb[5]; for (int l = 0; l < 5; ++l) featb[l] = allocf((size_t)B * Cs[l] * Ns[l]);
    float* mean   = allocf(1024);
    float* var    = allocf(1024);
    float* bstats = allocf(2 * B);
    int*   fidx   = alloci((size_t)B * 2048);
    int*   knn    = alloci((size_t)B * 2048 * KNN);
    float* mpts   = allocf((size_t)B * 2048 * 64);        // max B*S*D = 524288
    float* grp    = allocf((size_t)33554432);             // max B*S*2D*32 (col-padded)
    float* T1     = allocf((size_t)33554432);
    float* T2     = allocf((size_t)33554432);
    int*   idx3   = alloci((size_t)B * 4096 * 3);
    float* d3     = allocf((size_t)B * 4096 * 3);
    float* cat    = allocf((size_t)B * 1536 * 512);       // max B*(C1+C2)*N1
    float* upA    = allocf((size_t)B * 128 * 4096);
    float* upB    = allocf((size_t)B * 128 * 4096);

    // conv + BN(+res)(+relu): GEMM -> stats -> apply
    auto convbn = [&](const ConvP& c, const float* X, float* Yraw, float* out,
                      const float* res, int M, int Kd, int Nc, int ld, int nb, bool relu) {
        dim3 g((Nc + 31) / 32, (M + 15) / 16, nb);
        bool fast = (Kd % 32 == 0) && (M % 16 == 0) && (ld % 32 == 0);
        if (fast)
            k_gemm<false><<<g, 64, 0, stream>>>(c.w, X, Yraw, nullptr, M, Kd, Nc, ld, ld);
        else
            k_gemm<true ><<<g, 64, 0, stream>>>(c.w, X, Yraw, nullptr, M, Kd, Nc, ld, ld);
        k_bn_stats<<<M, 256, 0, stream>>>(Yraw, mean, var, M, Nc, ld, nb);
        int total = nb * M * Nc;
        k_bn_apply<<<(total + 255) / 256, 256, 0, stream>>>(
            out, Yraw, res, mean, var, c.g, c.b, M, Nc, ld, total, relu ? 1 : 0);
    };

    // ---- xyz (B,4096,3) + embed conv (9->64) ----
    {
        int tot = B * 4096 * 3;
        k_xyz0<<<(tot + 255) / 256, 256, 0, stream>>>(x, xyzb[0], tot);
    }
    convbn(embed, x, featb[0], featb[0], nullptr, 64, 9, 4096, 4096, B, true);

    // ---- encoder stages ----
    for (int i = 0; i < 4; ++i) {
        const int N = Ns[i], S = Ns[i + 1], D = Cs[i], C = 2 * D;

        k_fps<<<B, 256, 0, stream>>>(xyzb[i], fidx, N, S);
        { int tot = B * S * 3;
          k_gather_xyz<<<(tot + 255) / 256, 256, 0, stream>>>(xyzb[i], fidx, xyzb[i + 1], N, S, tot); }
        { int tot = B * S * D;
          k_gather_pts<<<(tot + 255) / 256, 256, 0, stream>>>(featb[i], fidx, mpts, N, S, D, tot); }
        { int tot = B * S;
          k_knn<<<(tot + 63) / 64, 64, 0, stream>>>(xyzb[i + 1], xyzb[i], knn, N, S, tot); }

        k_zero<<<1, 32, 0, stream>>>(bstats, 2 * B);
        { dim3 g(256, B);
          k_group_stats<<<g, 256, 0, stream>>>(featb[i], mpts, knn, bstats, N, S, D); }
        { int tot = B * S * KNN * D;
          k_group_write<<<(tot + 255) / 256, 256, 0, stream>>>(
              featb[i], mpts, knn, bstats, st[i].alpha, st[i].beta, grp, N, S, D, tot); }

        // pre-extraction: transfer + 2 res blocks over (B*S, C, 24; ld=32)
        convbn(st[i].transfer, grp, T1, T1, nullptr, C, C, KNN, 32, B * S, true);
        convbn(st[i].pre[0].n1, T1, T2, T2, nullptr, C, C, KNN, 32, B * S, true);
        convbn(st[i].pre[0].n2, T2, grp, grp, T1, C, C, KNN, 32, B * S, true);
        convbn(st[i].pre[1].n1, grp, T1, T1, nullptr, C, C, KNN, 32, B * S, true);
        convbn(st[i].pre[1].n2, T1, T2, T2, grp, C, C, KNN, 32, B * S, true);

        { int tot = B * S * C;
          k_maxpool<<<(tot + 255) / 256, 256, 0, stream>>>(T2, featb[i + 1], S, C, tot); }

        // pos res blocks over (B, C, S)
        convbn(st[i].pos[0].n1, featb[i + 1], T1, T1, nullptr, C, C, S, S, B, true);
        convbn(st[i].pos[0].n2, T1, T2, featb[i + 1], featb[i + 1], C, C, S, S, B, true);
        convbn(st[i].pos[1].n1, featb[i + 1], T1, T1, nullptr, C, C, S, S, B, true);
        convbn(st[i].pos[1].n2, T1, T2, featb[i + 1], featb[i + 1], C, C, S, S, B, true);
    }

    // ---- feature propagation (decoder) ----
    const int dec[4] = {512, 256, 128, 128};
    float* up = featb[4];
    int C2 = 1024;
    for (int j = 0; j < 4; ++j) {
        const int ci = 4 - j;
        const int N1 = Ns[ci - 1], N2 = Ns[ci], C1 = Cs[ci - 1], OC = dec[j];
        const int Ct = C1 + C2;

        { int tot = B * N1;
          k_nn3<<<(tot + 63) / 64, 64, 0, stream>>>(xyzb[ci - 1], xyzb[ci], idx3, d3, N1, N2, tot); }
        { int tot = B * C1 * N1;
          k_copychan<<<(tot + 255) / 256, 256, 0, stream>>>(featb[ci - 1], cat, C1, Ct, N1, tot); }
        { int tot = B * C2 * N1;
          k_interp<<<(tot + 255) / 256, 256, 0, stream>>>(up, idx3, d3, cat, C1, C2, Ct, N1, N2, tot); }

        convbn(fp[j].fuse, cat, T1, T1, nullptr, OC, Ct, N1, N1, B, true);
        convbn(fp[j].ext.n1, T1, T2, T2, nullptr, OC, OC, N1, N1, B, true);
        float* next = (j & 1) ? upB : upA;
        convbn(fp[j].ext.n2, T2, grp, next, T1, OC, OC, N1, N1, B, true);
        up = next;
        C2 = OC;
    }

    // ---- classifier head ----
    convbn(cls1, up, T1, T1, nullptr, 128, 128, 4096, 4096, B, true);
    {
        dim3 g((4096 + 31) / 32, 1, B);
        k_gemm<true><<<g, 64, 0, stream>>>(clsW, T1, (float*)d_out, clsB, 13, 128, 4096, 4096, 4096);
    }
}
